// RestorationNet_4724464026082
// MI455X (gfx1250) — compile-verified, matrix operand order checked by reading, not run on hardware
//
#include <hip/hip_runtime.h>
#include <hip/hip_bf16.h>

typedef __attribute__((ext_vector_type(16))) _Float16 v16h;
typedef __attribute__((ext_vector_type(2)))  _Float16 h2v;
typedef __attribute__((ext_vector_type(8)))  float    v8f;

union Frag { v16h v; h2v p[8]; };

static constexpr int BATCH = 2;

// ---------------------------------------------------------------------------
// f32 -> f16 convert (plain, for DCN weights: [n][ci*9+r] order kept)
// ---------------------------------------------------------------------------
__global__ void k_cvt16(const float* __restrict__ s, _Float16* __restrict__ d, int n) {
  int i = blockIdx.x * blockDim.x + threadIdx.x;
  if (i < n) d[i] = (_Float16)s[i];
}

// ---------------------------------------------------------------------------
// f32 -> f16 convert with tap-major permute: dst[n][r*Cin+ci] = src[n][ci][r]
// ---------------------------------------------------------------------------
__global__ void k_cvt_perm(const float* __restrict__ s, _Float16* __restrict__ d,
                           int Cin, int total) {
  int i = blockIdx.x * blockDim.x + threadIdx.x;
  if (i >= total) return;
  int r = i % 9; int t = i / 9; int ci = t % Cin; int n = t / Cin;
  d[(size_t)n * Cin * 9 + r * Cin + ci] = (_Float16)s[i];
}

// ---------------------------------------------------------------------------
// first conv weight: (64,3,3,3) -> rows padded to 32 (K = ci*9+r, zeros 27..31)
// ---------------------------------------------------------------------------
__global__ void k_cvt_pad27(const float* __restrict__ s, _Float16* __restrict__ d,
                            int total) {
  int i = blockIdx.x * blockDim.x + threadIdx.x;
  if (i >= total) return;
  int K = i & 31; int n = i >> 5;
  d[i] = (K < 27) ? (_Float16)s[n * 27 + K] : (_Float16)0.f;
}

// ---------------------------------------------------------------------------
// im2col for the 3-channel first conv: col[b][HW][32], K = ci*9+r (pad to 32)
// ---------------------------------------------------------------------------
__global__ void k_im2col3(const float* __restrict__ x, _Float16* __restrict__ col,
                          int H, int W, int total) {
  int i = blockIdx.x * blockDim.x + threadIdx.x;
  if (i >= total) return;
  int K = i & 31; int t = i >> 5; int p = t % (H * W); int b = t / (H * W);
  _Float16 v = (_Float16)0.f;
  if (K < 27) {
    int ci = K / 9, r = K - ci * 9;
    int h = p / W + r / 3 - 1, w = p % W + (r % 3) - 1;
    if ((unsigned)h < (unsigned)H && (unsigned)w < (unsigned)W)
      v = (_Float16)x[((size_t)b * 3 + ci) * H * W + h * W + w];
  }
  col[i] = v;
}

// ---------------------------------------------------------------------------
// pack f32 NCHW tensor into channel-last f16 buffer [b][HW][Ctot] at chan c0
// ---------------------------------------------------------------------------
__global__ void k_pack16(const float* __restrict__ s, _Float16* __restrict__ d,
                         int C, int HW, int Ctot, int c0, int total) {
  int i = blockIdx.x * blockDim.x + threadIdx.x;
  if (i >= total) return;
  int p = i % HW; int t = i / HW; int c = t % C; int b = t / C;
  d[((size_t)b * HW + p) * Ctot + c0 + c] = (_Float16)s[i];
}

// ---------------------------------------------------------------------------
// out = a (+ b), optional f32 NCHW store and/or channel-last f16 store
// ---------------------------------------------------------------------------
__global__ void k_add(const float* __restrict__ a, const float* __restrict__ bsrc,
                      int C, int HW, float* __restrict__ o32,
                      _Float16* __restrict__ o16, int Ctot, int c0, int total) {
  int i = blockIdx.x * blockDim.x + threadIdx.x;
  if (i >= total) return;
  float v = a[i] + (bsrc ? bsrc[i] : 0.f);
  if (o32) o32[i] = v;
  if (o16) {
    int p = i % HW; int t = i / HW; int c = t % C; int b = t / C;
    o16[((size_t)b * HW + p) * Ctot + c0 + c] = (_Float16)v;
  }
}

// ---------------------------------------------------------------------------
// bilinear 4x upsample of (B,3,H,W) -> (B,3,4H,4W), half-pixel centers
// ---------------------------------------------------------------------------
__global__ void k_resize4(const float* __restrict__ x, float* __restrict__ o,
                          int H, int W, int total) {
  int i = blockIdx.x * blockDim.x + threadIdx.x;
  if (i >= total) return;
  int OW = 4 * W, OH = 4 * H;
  int ox = i % OW; int t = i / OW; int oy = t % OH; t /= OH; int c = t % 3; int b = t / 3;
  float sy = fminf(fmaxf((oy + 0.5f) * 0.25f - 0.5f, 0.f), (float)(H - 1));
  float sx = fminf(fmaxf((ox + 0.5f) * 0.25f - 0.5f, 0.f), (float)(W - 1));
  int y0 = (int)sy, x0 = (int)sx;
  int y1 = min(y0 + 1, H - 1), x1 = min(x0 + 1, W - 1);
  float fy = sy - y0, fx = sx - x0;
  const float* xb = x + ((size_t)b * 3 + c) * H * W;
  o[i] = (1.f - fy) * ((1.f - fx) * xb[y0 * W + x0] + fx * xb[y0 * W + x1])
       +        fy  * ((1.f - fx) * xb[y1 * W + x0] + fx * xb[y1 * W + x1]);
}

// ---------------------------------------------------------------------------
// pixel shuffle r=2 + leaky relu: (B,4C,H,W) f32 -> (B,C,2H,2W) f32
// ---------------------------------------------------------------------------
__global__ void k_pixshuf(const float* __restrict__ in, int C, int H, int W,
                          float* __restrict__ o32, int total) {
  int i = blockIdx.x * blockDim.x + threadIdx.x;
  if (i >= total) return;
  int OW = 2 * W, OH = 2 * H;
  int ox = i % OW; int t = i / OW; int oy = t % OH; t /= OH; int c = t % C; int b = t / C;
  int ii = oy & 1, jj = ox & 1;
  float v = in[((size_t)b * 4 * C + c * 4 + ii * 2 + jj) * H * W + (oy >> 1) * W + (ox >> 1)];
  o32[i] = v > 0.f ? v : 0.1f * v;
}

// ---------------------------------------------------------------------------
// Modulated-deformable sampling -> col[b][HW][C*9] (f16), K = ci*9+k.
// ref is channel-last [b][HW][refCtot]: per-corner reads are unit-stride in c.
// ---------------------------------------------------------------------------
__global__ void k_dcn_sample(const float* __restrict__ om, const float* __restrict__ po,
                             const _Float16* __restrict__ ref, int refCtot, int refc0,
                             _Float16* __restrict__ col, int C, int H, int W, int total) {
  int i = blockIdx.x * blockDim.x + threadIdx.x;
  if (i >= total) return;
  int HW = H * W;
  int p = i % HW; int t = i / HW;
  int k = t % 9; t /= 9;
  int g = t % 8; int b = t / 8;
  int h = p / W, w = p % W;
  const float* omB = om + (size_t)b * 216 * HW;
  float dy = omB[(g * 18 + 2 * k) * HW + p]     + po[(((size_t)b * 9 + k) * HW + p) * 2 + 1];
  float dx = omB[(g * 18 + 2 * k + 1) * HW + p] + po[(((size_t)b * 9 + k) * HW + p) * 2 + 0];
  float m = 1.f / (1.f + expf(-omB[(144 + g * 9 + k) * HW + p]));
  float py = (float)h + (float)(k / 3 - 1) + dy;
  float px = (float)w + (float)(k % 3 - 1) + dx;
  float y0f = floorf(py), x0f = floorf(px);
  float wy = py - y0f, wx = px - x0f;
  int y0 = (int)y0f, x0 = (int)x0f, y1 = y0 + 1, x1 = x0 + 1;
  float w00 = ((y0 >= 0 && y0 < H && x0 >= 0 && x0 < W) ? (1.f - wy) * (1.f - wx) : 0.f) * m;
  float w01 = ((y0 >= 0 && y0 < H && x1 >= 0 && x1 < W) ? (1.f - wy) * wx : 0.f) * m;
  float w10 = ((y1 >= 0 && y1 < H && x0 >= 0 && x0 < W) ? wy * (1.f - wx) : 0.f) * m;
  float w11 = ((y1 >= 0 && y1 < H && x1 >= 0 && x1 < W) ? wy * wx : 0.f) * m;
  int cy0 = min(max(y0, 0), H - 1), cy1 = min(max(y1, 0), H - 1);
  int cx0 = min(max(x0, 0), W - 1), cx1 = min(max(x1, 0), W - 1);
  int Cg = C >> 3;
  int Ktot = C * 9;
  const size_t bHW = (size_t)b * HW;
  const int cbase = refc0 + g * Cg;
  const _Float16* r00 = ref + (bHW + cy0 * W + cx0) * refCtot + cbase;
  const _Float16* r01 = ref + (bHW + cy0 * W + cx1) * refCtot + cbase;
  const _Float16* r10 = ref + (bHW + cy1 * W + cx0) * refCtot + cbase;
  const _Float16* r11 = ref + (bHW + cy1 * W + cx1) * refCtot + cbase;
  _Float16* colB = col + (bHW + p) * Ktot + (size_t)g * Cg * 9 + k;
  for (int c = 0; c < Cg; ++c) {
    float v = w00 * (float)r00[c] + w01 * (float)r01[c]
            + w10 * (float)r10[c] + w11 * (float)r11[c];
    colB[c * 9] = (_Float16)v;
  }
}

// ---------------------------------------------------------------------------
// fused GEMM epilogue: +bias, +residual(NCHW f32), ReLU/LReLU,
// f32 NCHW store and/or channel-last f16 store
// ---------------------------------------------------------------------------
__device__ __forceinline__ void store_tile(const v8f& acc, int b, int n, int mbase,
                                           bool hi, int HW, int Cout, float bb,
                                           int actmode, const float* resid,
                                           float* out32, _Float16* out16,
                                           int oCtot, int oc0) {
  if (n >= Cout) return;
#pragma unroll
  for (int j = 0; j < 8; ++j) {
    int pp = mbase + j + (hi ? 8 : 0);
    if (pp >= HW) continue;
    float v = acc[j] + bb;
    if (resid) v += resid[((size_t)b * Cout + n) * HW + pp];
    if (actmode == 1) v = fmaxf(v, 0.f);
    else if (actmode == 2) v = v > 0.f ? v : 0.1f * v;
    if (out32) out32[((size_t)b * Cout + n) * HW + pp] = v;
    if (out16) out16[((size_t)b * HW + pp) * oCtot + oc0 + n] = (_Float16)v;
  }
}

#define WMMA_F16(A, B, C) \
  __builtin_amdgcn_wmma_f32_16x16x32_f16(false, A, false, B, (short)0, C, false, false)

// ---------------------------------------------------------------------------
// Implicit-GEMM 3x3 conv (pad 1), tap-major reduction K' = r*Cin + ci.
// Activations are channel-last f16 [b][HW][Cin] (Cin % 32 == 0), so every
// A fragment is 2x global_load_b128 (unit-stride K), like the weight side.
// Wave computes 32 pixels x 32 channels (2x2 blocking, 4 v_wmma per chunk).
// ---------------------------------------------------------------------------
__global__ __launch_bounds__(64)
void k_conv3x3_wmma(const _Float16* __restrict__ act, int Cin, int H, int W,
                    const _Float16* __restrict__ wgt, const float* __restrict__ bias,
                    int Cout, int actmode, const float* __restrict__ resid,
                    float* __restrict__ out32, _Float16* __restrict__ out16,
                    int oCtot, int oc0) {
  const int lane = threadIdx.x;
  const int mrow = lane & 15;
  const bool hi = lane >= 16;
  const int khb = hi ? 8 : 0;
  const int HW = H * W;
  const int Ktot = Cin * 9;
  const int mbase = blockIdx.x * 32;
  const int nbase = (blockIdx.y * 2 + threadIdx.y) * 32;
  const int b = blockIdx.z;
  const int p0 = mbase + mrow, p1 = p0 + 16;
  const int n0 = nbase + mrow, n1 = n0 + 16;
  const int oh0 = p0 / W, ow0 = p0 - oh0 * W;
  const int oh1 = p1 / W, ow1 = p1 - oh1 * W;
  const bool p0ok = p0 < HW, p1ok = p1 < HW;
  const _Float16* actB = act + (size_t)b * HW * Cin;
  const _Float16* w0 = wgt + (size_t)(n0 < Cout ? n0 : 0) * Ktot;
  const _Float16* w1 = wgt + (size_t)(n1 < Cout ? n1 : 0) * Ktot;

  v8f c00 = {}, c01 = {}, c10 = {}, c11 = {};
#pragma unroll
  for (int r = 0; r < 9; ++r) {
    const int ky = r / 3 - 1, kx = (r % 3) - 1;
    const int ih0 = oh0 + ky, iw0 = ow0 + kx;
    const int ih1 = oh1 + ky, iw1 = ow1 + kx;
    const bool v0 = p0ok && (unsigned)ih0 < (unsigned)H && (unsigned)iw0 < (unsigned)W;
    const bool v1 = p1ok && (unsigned)ih1 < (unsigned)H && (unsigned)iw1 < (unsigned)W;
    const int off0 = v0 ? (ih0 * W + iw0) : 0;
    const int off1 = v1 ? (ih1 * W + iw1) : 0;
    const _Float16* pa0 = actB + (size_t)off0 * Cin;
    const _Float16* pa1 = actB + (size_t)off1 * Cin;
    const _Float16* wr0 = w0 + r * Cin;
    const _Float16* wr1 = w1 + r * Cin;
    for (int cb = 0; cb < Cin; cb += 32) {
      __builtin_prefetch(wr0 + cb + 128, 0, 1);
      Frag a0, a1, b0, b1;
#pragma unroll
      for (int d = 0; d < 8; ++d) {
        const int kk = cb + ((d < 4) ? 2 * d : 2 * d + 8) + khb;
        b0.p[d] = *(const h2v*)(wr0 + kk);
        b1.p[d] = *(const h2v*)(wr1 + kk);
      }
      if (v0) {
#pragma unroll
        for (int d = 0; d < 8; ++d) {
          const int kk = cb + ((d < 4) ? 2 * d : 2 * d + 8) + khb;
          a0.p[d] = *(const h2v*)(pa0 + kk);
        }
      } else {
#pragma unroll
        for (int h = 0; h < 16; ++h) a0.v[h] = (_Float16)0.f;
      }
      if (v1) {
#pragma unroll
        for (int d = 0; d < 8; ++d) {
          const int kk = cb + ((d < 4) ? 2 * d : 2 * d + 8) + khb;
          a1.p[d] = *(const h2v*)(pa1 + kk);
        }
      } else {
#pragma unroll
        for (int h = 0; h < 16; ++h) a1.v[h] = (_Float16)0.f;
      }
      c00 = WMMA_F16(a0.v, b0.v, c00);
      c01 = WMMA_F16(a0.v, b1.v, c01);
      c10 = WMMA_F16(a1.v, b0.v, c10);
      c11 = WMMA_F16(a1.v, b1.v, c11);
    }
  }
  const float bb0 = (n0 < Cout) ? bias[n0] : 0.f;
  const float bb1 = (n1 < Cout) ? bias[n1] : 0.f;
  store_tile(c00, b, n0, mbase,      hi, HW, Cout, bb0, actmode, resid, out32, out16, oCtot, oc0);
  store_tile(c01, b, n1, mbase,      hi, HW, Cout, bb1, actmode, resid, out32, out16, oCtot, oc0);
  store_tile(c10, b, n0, mbase + 16, hi, HW, Cout, bb0, actmode, resid, out32, out16, oCtot, oc0);
  store_tile(c11, b, n1, mbase + 16, hi, HW, Cout, bb1, actmode, resid, out32, out16, oCtot, oc0);
}

// ---------------------------------------------------------------------------
// Dense GEMM: col[b][M][Ktot] (f16) x wgt[Cout][Ktot] (f16), Ktot % 32 == 0.
// ---------------------------------------------------------------------------
__global__ __launch_bounds__(64)
void k_gemm_wmma(const _Float16* __restrict__ col, int M, int Ktot,
                 const _Float16* __restrict__ wgt, const float* __restrict__ bias,
                 int Cout, int actmode, const float* __restrict__ resid,
                 float* __restrict__ out32, _Float16* __restrict__ out16,
                 int oCtot, int oc0) {
  const int lane = threadIdx.x;
  const int mrow = lane & 15;
  const bool hi = lane >= 16;
  const int khb = hi ? 8 : 0;
  const int mbase = blockIdx.x * 32;
  const int nbase = (blockIdx.y * 2 + threadIdx.y) * 32;
  const int b = blockIdx.z;
  const int p0 = mbase + mrow, p1 = p0 + 16;
  const int n0 = nbase + mrow, n1 = n0 + 16;
  const bool p0ok = p0 < M, p1ok = p1 < M;
  const _Float16* a0p = col + ((size_t)b * M + (p0ok ? p0 : 0)) * Ktot;
  const _Float16* a1p = col + ((size_t)b * M + (p1ok ? p1 : 0)) * Ktot;
  const _Float16* w0 = wgt + (size_t)(n0 < Cout ? n0 : 0) * Ktot;
  const _Float16* w1 = wgt + (size_t)(n1 < Cout ? n1 : 0) * Ktot;

  v8f c00 = {}, c01 = {}, c10 = {}, c11 = {};
  for (int k0 = 0; k0 < Ktot; k0 += 32) {
    __builtin_prefetch(w0 + k0 + 128, 0, 1);
    Frag a0, a1, b0, b1;
#pragma unroll
    for (int d = 0; d < 8; ++d) {
      const int kk = k0 + ((d < 4) ? 2 * d : 2 * d + 8) + khb;
      a0.p[d] = *(const h2v*)(a0p + kk);
      a1.p[d] = *(const h2v*)(a1p + kk);
      b0.p[d] = *(const h2v*)(w0 + kk);
      b1.p[d] = *(const h2v*)(w1 + kk);
    }
    c00 = WMMA_F16(a0.v, b0.v, c00);
    c01 = WMMA_F16(a0.v, b1.v, c01);
    c10 = WMMA_F16(a1.v, b0.v, c10);
    c11 = WMMA_F16(a1.v, b1.v, c11);
  }
  const float bb0 = (n0 < Cout) ? bias[n0] : 0.f;
  const float bb1 = (n1 < Cout) ? bias[n1] : 0.f;
  store_tile(c00, b, n0, mbase,      hi, M, Cout, bb0, actmode, resid, out32, out16, oCtot, oc0);
  store_tile(c01, b, n1, mbase,      hi, M, Cout, bb1, actmode, resid, out32, out16, oCtot, oc0);
  store_tile(c10, b, n0, mbase + 16, hi, M, Cout, bb0, actmode, resid, out32, out16, oCtot, oc0);
  store_tile(c11, b, n1, mbase + 16, hi, M, Cout, bb1, actmode, resid, out32, out16, oCtot, oc0);
}

// ===========================================================================
// Host-side orchestration
// ===========================================================================
namespace {

struct Bufs {
  _Float16 *cat, *offA, *offB, *col, *h16a, *h16b, *t16, *tailin, *t1;
  float *om, *h32a, *h32b, *ts, *fin, *base, *x0a, *x0b;
};

struct ScaleW {
  const _Float16 *oc1, *oc2, *om, *dw, *head, *bw1, *bw2, *tail1, *tail2;
  const float *oc1b, *oc2b, *omb, *db, *headb, *bb1, *bb2, *tail1b, *tail2b;
};

inline void conv(hipStream_t st, const _Float16* act, int Cin, int H, int W,
                 const _Float16* w, const float* b, int Cout, int actmode,
                 const float* resid, float* o32, _Float16* o16, int oCtot, int oc0) {
  dim3 blk(32, 2, 1);
  dim3 grd((H * W + 31) / 32, (Cout + 63) / 64, BATCH);
  k_conv3x3_wmma<<<grd, blk, 0, st>>>(act, Cin, H, W, w, b, Cout, actmode, resid,
                                      o32, o16, oCtot, oc0);
}

inline void gemm(hipStream_t st, const _Float16* col, int M, int Ktot,
                 const _Float16* w, const float* b, int Cout, int actmode,
                 const float* resid, float* o32, _Float16* o16, int oCtot, int oc0) {
  dim3 blk(32, 2, 1);
  dim3 grd((M + 31) / 32, (Cout + 63) / 64, BATCH);
  k_gemm_wmma<<<grd, blk, 0, st>>>(col, M, Ktot, w, b, Cout, actmode, resid,
                                   o32, o16, oCtot, oc0);
}

inline void ew(int total, dim3& g, dim3& b) {
  b = dim3(256, 1, 1);
  g = dim3((total + 255) / 256, 1, 1);
}

void run_scale(hipStream_t st, const Bufs& Q, int C, int H, int W,
               const float* rf, const float* po, const ScaleW& P, bool final_,
               const float* x0in, float* x0out, float* d_outF) {
  const int HW = H * W;
  const int Ctot = 64 + C;
  dim3 g, blk;

  // concat(x0, rf) -> cat (channel-last f16)
  ew(BATCH * C * HW, g, blk);
  k_pack16<<<g, blk, 0, st>>>(rf, Q.cat, C, HW, Ctot, 64, BATCH * C * HW);
  ew(BATCH * 64 * HW, g, blk);
  k_pack16<<<g, blk, 0, st>>>(x0in, Q.cat, 64, HW, Ctot, 0, BATCH * 64 * HW);

  // offset branch: two lrelu convs, then om conv (linear, f32)
  conv(st, Q.cat, Ctot, H, W, P.oc1, P.oc1b, C, 2, nullptr, nullptr, Q.offA, C, 0);
  conv(st, Q.offA, C, H, W, P.oc2, P.oc2b, C, 2, nullptr, nullptr, Q.offB, C, 0);
  conv(st, Q.offB, C, H, W, P.om, P.omb, 216, 0, nullptr, Q.om, nullptr, 0, 0);

  // deformable sampling -> col, then dense GEMM (+bias, lrelu) -> sw into cat[64:]
  int stotal = BATCH * 72 * HW;
  ew(stotal, g, blk);
  k_dcn_sample<<<g, blk, 0, st>>>(Q.om, po, Q.cat, Ctot, 64, Q.col, C, H, W, stotal);
  gemm(st, Q.col, HW, C * 9, P.dw, P.db, C, 2, nullptr, nullptr, Q.cat, Ctot, 64);

  // head conv on concat(x0, sw)
  conv(st, Q.cat, Ctot, H, W, P.head, P.headb, 64, 2, nullptr, Q.h32a, Q.h16a, 64, 0);

  // 16 residual blocks
  _Float16* h16c = Q.h16a; _Float16* h16n = Q.h16b;
  float* h32c = Q.h32a;    float* h32n = Q.h32b;
  for (int i = 0; i < 16; ++i) {
    const _Float16* w1 = P.bw1 + (size_t)i * 64 * 64 * 9;
    const _Float16* w2 = P.bw2 + (size_t)i * 64 * 64 * 9;
    conv(st, h16c, 64, H, W, w1, P.bb1 + i * 64, 64, 1, nullptr, nullptr, Q.t16, 64, 0);
    conv(st, Q.t16, 64, H, W, w2, P.bb2 + i * 64, 64, 0, h32c, h32n, h16n, 64, 0);
    _Float16* th = h16c; h16c = h16n; h16n = th;
    float* tf = h32c; h32c = h32n; h32n = tf;
  }

  // h + x0 -> tailin (f16)
  ew(BATCH * 64 * HW, g, blk);
  k_add<<<g, blk, 0, st>>>(h32c, x0in, 64, HW, nullptr, Q.tailin, 64, 0,
                           BATCH * 64 * HW);

  if (final_) {
    conv(st, Q.tailin, 64, H, W, P.tail1, P.tail1b, 32, 2, nullptr, nullptr, Q.t1, 32, 0);
    conv(st, Q.t1, 32, H, W, P.tail2, P.tail2b, 3, 0, nullptr, Q.fin, nullptr, 0, 0);
    ew(BATCH * 3 * HW, g, blk);
    k_add<<<g, blk, 0, st>>>(Q.fin, Q.base, 3, HW, d_outF, nullptr, 0, 0,
                             BATCH * 3 * HW);
  } else {
    conv(st, Q.tailin, 64, H, W, P.tail1, P.tail1b, 256, 0, nullptr, Q.ts, nullptr, 0, 0);
    int total = BATCH * 64 * (2 * H) * (2 * W);
    ew(total, g, blk);
    k_pixshuf<<<g, blk, 0, st>>>(Q.ts, 64, H, W, x0out, total);
  }
}

} // namespace

extern "C" void kernel_launch(void* const* d_in, const int* in_sizes, int n_in,
                              void* d_out, int out_size, void* d_ws, size_t ws_size,
                              hipStream_t stream) {
  (void)in_sizes; (void)n_in; (void)out_size; (void)ws_size;
  const float* X   = (const float*)d_in[0];
  const float* PO3 = (const float*)d_in[1];
  const float* PO2 = (const float*)d_in[2];
  const float* PO1 = (const float*)d_in[3];
  const float* RF3 = (const float*)d_in[4];
  const float* RF2 = (const float*)d_in[5];
  const float* RF1 = (const float*)d_in[6];
  // param leaves, jax pytree sorted-key order:
  // ce:7..12  large:13..30  medium:31..46  small:47..62
  auto F = [&](int i) { return (const float*)d_in[i]; };

  char* ws = (char*)d_ws;
  size_t cur = 0;
  auto bump = [&](size_t bytes) -> void* {
    void* p = ws + cur;
    cur += (bytes + 255) & ~(size_t)255;
    return p;
  };

  Bufs Q;
  Q.cat    = (_Float16*)bump((size_t)9437184 * 2);
  Q.offA   = (_Float16*)bump((size_t)4718592 * 2);
  Q.offB   = (_Float16*)bump((size_t)4718592 * 2);
  Q.col    = (_Float16*)bump((size_t)42467328 * 2);
  Q.h16a   = (_Float16*)bump((size_t)4718592 * 2);
  Q.h16b   = (_Float16*)bump((size_t)4718592 * 2);
  Q.t16    = (_Float16*)bump((size_t)4718592 * 2);
  Q.tailin = (_Float16*)bump((size_t)4718592 * 2);
  Q.t1     = (_Float16*)bump((size_t)2359296 * 2);
  Q.om     = (float*)bump((size_t)15925248 * 4);
  Q.h32a   = (float*)bump((size_t)4718592 * 4);
  Q.h32b   = (float*)bump((size_t)4718592 * 4);
  Q.ts     = (float*)bump((size_t)4718592 * 4);
  Q.fin    = (float*)bump((size_t)221184 * 4);
  Q.base   = (float*)bump((size_t)221184 * 4);
  Q.x0a    = (float*)bump((size_t)4718592 * 4);
  Q.x0b    = (float*)bump((size_t)4718592 * 4);

  // weight converts (every call; deterministic). Conv weights: tap-major permute.
  auto cwp = [&](int idx, int Cout, int Cin) -> _Float16* {
    size_t n = (size_t)Cout * Cin * 9;
    _Float16* d = (_Float16*)bump(n * 2);
    k_cvt_perm<<<(int)((n + 255) / 256), 256, 0, stream>>>(F(idx), d, Cin, (int)n);
    return d;
  };
  auto cw = [&](int idx, size_t n) -> _Float16* {   // plain (DCN dw)
    _Float16* d = (_Float16*)bump(n * 2);
    k_cvt16<<<(int)((n + 255) / 256), 256, 0, stream>>>(F(idx), d, (int)n);
    return d;
  };

  // first conv (Cin=3): padded rows of 32
  _Float16* ce_first = (_Float16*)bump((size_t)64 * 32 * 2);
  k_cvt_pad27<<<(64 * 32 + 255) / 256, 256, 0, stream>>>(F(12), ce_first, 64 * 32);

  auto cw_body = [&](int idx) -> _Float16* {   // (16,64,64,3,3): permute per block
    size_t n = (size_t)16 * 64 * 64 * 9;
    _Float16* d = (_Float16*)bump(n * 2);
    k_cvt_perm<<<(int)((n + 255) / 256), 256, 0, stream>>>(F(idx), d, 64, (int)n);
    return d;
  };

  _Float16* ce_bw1 = cw_body(9);
  _Float16* ce_bw2 = cw_body(10);

  ScaleW L{};  // large: 13..30
  L.bw1 = cw_body(15); L.bw2 = cw_body(16);
  L.dw = cw(18, (size_t)64 * 64 * 9);
  L.head = cwp(20, 64, 128);
  L.oc1 = cwp(22, 64, 128);  L.oc2 = cwp(24, 64, 64);
  L.om = cwp(26, 216, 64);
  L.tail1 = cwp(28, 32, 64); L.tail2 = cwp(30, 3, 32);
  L.bb1 = F(13); L.bb2 = F(14); L.db = F(17); L.headb = F(19);
  L.oc1b = F(21); L.oc2b = F(23); L.omb = F(25); L.tail1b = F(27); L.tail2b = F(29);

  ScaleW M{};  // medium: 31..46
  M.bw1 = cw_body(33); M.bw2 = cw_body(34);
  M.dw = cw(36, (size_t)128 * 128 * 9);
  M.head = cwp(38, 64, 192);
  M.oc1 = cwp(40, 128, 192); M.oc2 = cwp(42, 128, 128);
  M.om = cwp(44, 216, 128);  M.tail1 = cwp(46, 256, 64);
  M.tail2 = nullptr;
  M.bb1 = F(31); M.bb2 = F(32); M.db = F(35); M.headb = F(37);
  M.oc1b = F(39); M.oc2b = F(41); M.omb = F(43); M.tail1b = F(45); M.tail2b = nullptr;

  ScaleW S{};  // small: 47..62
  S.bw1 = cw_body(49); S.bw2 = cw_body(50);
  S.dw = cw(52, (size_t)256 * 256 * 9);
  S.head = cwp(54, 64, 320);
  S.oc1 = cwp(56, 256, 320); S.oc2 = cwp(58, 256, 256);
  S.om = cwp(60, 216, 256);  S.tail1 = cwp(62, 256, 64);
  S.tail2 = nullptr;
  S.bb1 = F(47); S.bb2 = F(48); S.db = F(51); S.headb = F(53);
  S.oc1b = F(55); S.oc2b = F(57); S.omb = F(59); S.tail1b = F(61); S.tail2b = nullptr;

  const int H0 = 48, W0 = 48, HW0 = H0 * W0;
  dim3 g, blk;

  // base = bilinear 4x upsample of x
  {
    int total = BATCH * 3 * 4 * H0 * 4 * W0;
    ew(total, g, blk);
    k_resize4<<<g, blk, 0, stream>>>(X, Q.base, H0, W0, total);
  }

  // context encoder: feat = res_body(lrelu(conv(x)))
  {
    int total = BATCH * HW0 * 32;
    ew(total, g, blk);
    k_im2col3<<<g, blk, 0, stream>>>(X, Q.col, H0, W0, total);
    gemm(stream, Q.col, HW0, 32, ce_first, F(11), 64, 2, nullptr, Q.h32a, Q.h16a, 64, 0);
    _Float16* h16c = Q.h16a; _Float16* h16n = Q.h16b;
    float* h32c = Q.h32a;    float* h32n = Q.h32b;
    for (int i = 0; i < 16; ++i) {
      conv(stream, h16c, 64, H0, W0, ce_bw1 + (size_t)i * 64 * 64 * 9, F(7) + i * 64,
           64, 1, nullptr, nullptr, Q.t16, 64, 0);
      conv(stream, Q.t16, 64, H0, W0, ce_bw2 + (size_t)i * 64 * 64 * 9, F(8) + i * 64,
           64, 0, h32c, h32n, h16n, 64, 0);
      _Float16* th = h16c; h16c = h16n; h16n = th;
      float* tf = h32c; h32c = h32n; h32n = tf;
    }
    int t2 = BATCH * 64 * HW0;
    ew(t2, g, blk);
    k_add<<<g, blk, 0, stream>>>(h32c, nullptr, 64, HW0, Q.x0a, nullptr, 0, 0, t2);
  }

  // three scale blocks: 48^2 (C=256) -> 96^2 (C=128) -> 192^2 (C=64, final)
  run_scale(stream, Q, 256, 48, 48,   RF3, PO3, S, false, Q.x0a, Q.x0b, nullptr);
  run_scale(stream, Q, 128, 96, 96,   RF2, PO2, M, false, Q.x0b, Q.x0a, nullptr);
  run_scale(stream, Q, 64, 192, 192,  RF1, PO1, L, true,  Q.x0a, nullptr, (float*)d_out);
}
